// PatchEmbedding_36953898614981
// MI455X (gfx1250) — matrix operand order repaired and checked by reading, not compiled, not run on hardware
//
#include <hip/hip_runtime.h>

// ---------------------------------------------------------------------------
// PatchEmbedding for MI455X (gfx1250, wave32, WMMA bf16).
//   x:[16,32,256,256] f32 -> 16 cells of [16,32,64,64]
//   per cell: silu(bn(conv3x3)) x2 + residual  -> enc
//   proj: out[b,n,o] = enc[n,b,:] . proj_w[o,:] + proj_b   (K=131072)
// Strategy: bf16 WMMA (16x16x32) with fp32 accumulate; BN folded into
// weights. Projection GEMM streams proj_w fp32 from HBM exactly once
// (402MB -> ~17us floor) converting to bf16 in-register, and is software-
// pipelined with register double-buffering so loads overlap WMMA.
// ---------------------------------------------------------------------------

typedef __bf16 bf16;
typedef __attribute__((ext_vector_type(16))) __bf16 v16bf;
typedef __attribute__((ext_vector_type(8)))  float  v8f;
typedef __attribute__((ext_vector_type(4)))  float  vf4;

union ABFrag { v16bf v; vf4 f[2]; };
union Pack8  { bf16 h[8]; vf4 f; };
union W16    { vf4 f[4]; float s[16]; };

__device__ __forceinline__ v8f wmma_bf16(v16bf a, v16bf b, v8f c) {
  // D = A(16x32 bf16) * B(32x16 bf16) + C(f32)
  return __builtin_amdgcn_wmma_f32_16x16x32_bf16(false, a, false, b,
                                                 (short)0, c, false, false);
}

// ---------------------------------------------------------------------------
// Pass 0: fold BN into conv weights; layout [n][tap][co][ci] bf16 so that a
// B-fragment lane (col co, 16 consecutive ci) is two contiguous b128 loads.
// ---------------------------------------------------------------------------
__global__ void prep_weights(const float* __restrict__ c1w, const float* __restrict__ g1,
                             const float* __restrict__ b1,  const float* __restrict__ m1,
                             const float* __restrict__ v1,
                             const float* __restrict__ c2w, const float* __restrict__ g2,
                             const float* __restrict__ b2,  const float* __restrict__ m2,
                             const float* __restrict__ v2,
                             bf16* __restrict__ wf1, bf16* __restrict__ wf2,
                             float* __restrict__ sh1, float* __restrict__ sh2) {
  int idx = blockIdx.x * 256 + threadIdx.x;
  if (idx >= 16 * 9 * 32 * 32) return;
  int ci = idx & 31; int t = idx >> 5;
  int co = t & 31;   t >>= 5;
  int tap = t % 9, n = t / 9;
  int nc = n * 32 + co;
  float s1 = g1[nc] / sqrtf(v1[nc] + 1e-5f);
  float s2 = g2[nc] / sqrtf(v2[nc] + 1e-5f);
  int src = (nc * 32 + ci) * 9 + tap;           // conv_w [n][co][ci][3][3]
  wf1[idx] = (bf16)(c1w[src] * s1);
  wf2[idx] = (bf16)(c2w[src] * s2);
  if (ci == 0 && tap == 0) {
    sh1[nc] = b1[nc] - m1[nc] * s1;
    sh2[nc] = b2[nc] - m2[nc] * s2;
  }
}

// ---------------------------------------------------------------------------
// Conv block as implicit GEMM. One workgroup = (cell n, batch b, 8-row strip).
// LDS tile: rows y0-1..y0+8, cols -1..64 (zero pad), NHWC with per-pixel
// channel stride CPAD=40 (80B -> 16B-aligned ds_load_b128, spread banks).
// Wave w (of 8) produces output row y0+w: 9 taps x 4 segs x 2 co-tiles WMMA.
// Output bf16 NCHW compact: [n][b][c][64*64]  (c-major == projection f-order)
// ---------------------------------------------------------------------------
#define CPAD 40
template <bool PASS2>
__global__ __launch_bounds__(256) void conv_block(
    const float* __restrict__ xin,   // full x NCHW (pass1 input / pass2 residual)
    const bf16*  __restrict__ hin,   // pass2 input  [n][b][c][4096]
    const bf16*  __restrict__ wfold, // [n][9][32][32]
    const float* __restrict__ shift, // [n][32]
    bf16* __restrict__ out)          // [n][b][c][4096]
{
  __shared__ bf16 tile[10 * 66 * CPAD];   // 52,800 B
  const int strip = blockIdx.x;           // 0..7
  const int b     = blockIdx.y;           // 0..15
  const int n     = blockIdx.z;           // 0..15
  const int rr = n >> 2, cc = n & 3;      // cell position in 4x4 grid
  const int y0 = strip * 8;
  const int tid = threadIdx.x;

  // ---- stage input tile into LDS (zero-padded halo), NCHW -> NHWC ----------
  for (int e = tid; e < 66 * 10 * 32; e += 256) {
    int xcol = e % 66; int t = e / 66;
    int yy = t % 10;   int ci = t / 10;
    int gy = y0 + yy - 1, gx = xcol - 1;
    bf16 bval = (bf16)0.0f;
    if (gy >= 0 && gy < 64 && gx >= 0 && gx < 64) {
      if (PASS2)
        bval = hin[((size_t)((n * 16 + b) * 32 + ci) << 12) + gy * 64 + gx];
      else
        bval = (bf16)xin[((size_t)(b * 32 + ci) << 16) + (rr * 64 + gy) * 256 + (cc * 64 + gx)];
    }
    tile[(yy * 66 + xcol) * CPAD + ci] = bval;
  }
  __syncthreads();

  const int wave = tid >> 5;
  const int lane = tid & 31;
  const int mrow = lane & 15;   // A row / B column (co)
  const int half = lane >> 4;   // K-half selector per ISA 16-bit layouts
  const int y = y0 + wave;

  v8f acc[4][2] = {};
  const bf16* wn = wfold + n * 9 * 1024;

  #pragma unroll
  for (int tap = 0; tap < 9; ++tap) {
    const int dy = tap / 3, dx = tap % 3;
    // B fragments: lane holds column co, K rows half*16..+15 (contiguous ci)
    ABFrag B0, B1;
    const bf16* wp = wn + tap * 1024 + half * 16;
    B0.f[0] = *(const vf4*)(wp + mrow * 32);
    B0.f[1] = *(const vf4*)(wp + mrow * 32 + 8);
    B1.f[0] = *(const vf4*)(wp + (mrow + 16) * 32);
    B1.f[1] = *(const vf4*)(wp + (mrow + 16) * 32 + 8);
    #pragma unroll
    for (int seg = 0; seg < 4; ++seg) {
      // A fragment: lane<16 -> K{0..15}; lane>=16 -> K{16..31}
      const bf16* ap = &tile[((wave + dy) * 66 + seg * 16 + mrow + dx) * CPAD];
      ABFrag A;
      A.f[0] = *(const vf4*)(ap + half * 16);
      A.f[1] = *(const vf4*)(ap + half * 16 + 8);
      acc[seg][0] = wmma_bf16(A.v, B0.v, acc[seg][0]);
      acc[seg][1] = wmma_bf16(A.v, B1.v, acc[seg][1]);
    }
  }

  // ---- epilogue: +shift, SiLU, (+residual), pack 8 x-consecutive bf16 ------
  #pragma unroll
  for (int nt = 0; nt < 2; ++nt) {
    const int co = mrow + nt * 16;
    const float sh = shift[n * 32 + co];
    #pragma unroll
    for (int seg = 0; seg < 4; ++seg) {
      const int xb = seg * 16 + half * 8;       // 8 consecutive x per lane
      vf4 r0 = {}, r1 = {};
      if (PASS2) {
        const float* xr = xin + ((size_t)(b * 32 + co) << 16)
                          + (rr * 64 + y) * 256 + (cc * 64 + xb);
        r0 = *(const vf4*)xr;
        r1 = *(const vf4*)(xr + 4);
      }
      Pack8 p;
      #pragma unroll
      for (int r = 0; r < 8; ++r) {
        float s = acc[seg][nt][r] + sh;
        float v = s / (1.0f + __builtin_expf(-s));   // SiLU
        if (PASS2) v += (r < 4) ? r0[r] : r1[r - 4]; // residual + p
        p.h[r] = (bf16)v;
      }
      *(vf4*)(out + ((size_t)((n * 16 + b) * 32 + co) << 12) + y * 64 + xb) = p.f;
    }
  }
}

// ---------------------------------------------------------------------------
// Projection GEMM helpers: B fragment = 16 consecutive f of one proj_w row,
// loaded fp32 (4x b128) and converted in-register; A fragments = 16 M-tiles
// (M-tile == batch, lane-row == cell) of enc bf16, two b128 each.
// ---------------------------------------------------------------------------
__device__ __forceinline__ v16bf load_wfrag(const float* __restrict__ wp) {
  W16 wl;
  wl.f[0] = *(const vf4*)(wp);
  wl.f[1] = *(const vf4*)(wp + 4);
  wl.f[2] = *(const vf4*)(wp + 8);
  wl.f[3] = *(const vf4*)(wp + 12);
  v16bf bv;
  #pragma unroll
  for (int j = 0; j < 16; ++j) bv[j] = (bf16)wl.s[j];
  return bv;
}

__device__ __forceinline__ void load_afrags(ABFrag* A, const bf16* __restrict__ p) {
  #pragma unroll
  for (int mt = 0; mt < 16; ++mt) {
    const bf16* ap = p + ((size_t)mt << 17);
    A[mt].f[0] = *(const vf4*)(ap);
    A[mt].f[1] = *(const vf4*)(ap + 8);
  }
}

__device__ __forceinline__ void do_wmma16(v8f* acc, const ABFrag* A, v16bf B) {
  #pragma unroll
  for (int mt = 0; mt < 16; ++mt)
    acc[mt] = wmma_bf16(A[mt].v, B, acc[mt]);
}

// ---------------------------------------------------------------------------
// Projection GEMM: out rows r=b*16+cell (M=256), cols o (N=768), K=131072.
// Wave keeps all 16 M-tile accumulators (128 VGPRs) so each proj_w fragment
// feeds 16 WMMAs -> proj_w read from HBM exactly once; enc (64MB) stays L2-
// resident across its 48 re-reads. 32-way K split -> deterministic partials.
// K loop is software-pipelined: ping-pong A/B register buffers (~440 VGPRs,
// fine out of 1024/wave) so the 36 loads of step s+1 fly under step s's WMMAs.
// ---------------------------------------------------------------------------
__global__ __launch_bounds__(128) void proj_gemm(
    const bf16*  __restrict__ enc,     // [n][b][131072] bf16
    const float* __restrict__ pw,      // [768][131072] f32
    float* __restrict__ partial)       // [32][256][768] f32
{
  const int ot   = blockIdx.x;               // 0..47 N-tiles
  const int wave = threadIdx.x >> 5;
  const int lane = threadIdx.x & 31;
  const int ks   = blockIdx.y * 4 + wave;    // 0..31 K-split
  const int mrow = lane & 15;
  const int half = lane >> 4;
  const int o    = ot * 16 + mrow;
  const size_t fstart = (size_t)ks * 4096;   // 131072/32
  const float* wbase = pw + (size_t)o * 131072 + fstart + half * 16;
  const bf16*  abase = enc + ((size_t)(mrow * 16) << 17) + fstart + half * 16;

  v8f acc[16] = {};
  ABFrag A0[16], A1[16];
  v16bf B0, B1;

  // prologue: stage step 0
  B0 = load_wfrag(wbase);
  load_afrags(A0, abase);

  for (int base = 0; base < 128; base += 2) {
    const int s1 = base + 1;                         // always < 128
    // stage step base+1 while computing step base
    B1 = load_wfrag(wbase + s1 * 32);
    load_afrags(A1, abase + (size_t)s1 * 32);
    __builtin_prefetch(wbase + (base + 16) * 32, 0, 1);  // weight stream ahead
    do_wmma16(acc, A0, B0);
    // stage step base+2 (clamped in-bounds dummy on last iter) under step base+1
    const int s2 = (base + 2 < 128) ? base + 2 : 0;
    B0 = load_wfrag(wbase + s2 * 32);
    load_afrags(A0, abase + (size_t)s2 * 32);
    do_wmma16(acc, A1, B1);
  }

  #pragma unroll
  for (int mt = 0; mt < 16; ++mt) {
    #pragma unroll
    for (int r = 0; r < 8; ++r) {
      const int row = mt * 16 + half * 8 + r;  // = b*16 + cell
      partial[((size_t)ks * 256 + row) * 768 + o] = acc[mt][r];
    }
  }
}

__global__ void reduce_bias(const float* __restrict__ partial,
                            const float* __restrict__ pb,
                            float* __restrict__ outp) {
  int idx = blockIdx.x * 256 + threadIdx.x;
  if (idx >= 256 * 768) return;
  int o = idx % 768;
  float s = pb[o];
  #pragma unroll
  for (int ks = 0; ks < 32; ++ks) s += partial[(size_t)ks * 196608 + idx];
  outp[idx] = s;
}

// ---------------------------------------------------------------------------
extern "C" void kernel_launch(void* const* d_in, const int* in_sizes, int n_in,
                              void* d_out, int out_size, void* d_ws, size_t ws_size,
                              hipStream_t stream) {
  const float* x   = (const float*)d_in[0];
  const float* c1w = (const float*)d_in[1];
  const float* g1  = (const float*)d_in[2];
  const float* b1  = (const float*)d_in[3];
  const float* m1  = (const float*)d_in[4];
  const float* v1  = (const float*)d_in[5];
  const float* c2w = (const float*)d_in[6];
  const float* g2  = (const float*)d_in[7];
  const float* b2  = (const float*)d_in[8];
  const float* m2  = (const float*)d_in[9];
  const float* v2  = (const float*)d_in[10];
  const float* pw  = (const float*)d_in[11];
  const float* pb  = (const float*)d_in[12];

  char* ws = (char*)d_ws;                      // needs ~160 MB
  bf16*  wf1 = (bf16*)(ws);                    // 294,912 B
  bf16*  wf2 = (bf16*)(ws + 294912);           // 294,912 B
  float* sh1 = (float*)(ws + 589824);          // 2,048 B
  float* sh2 = (float*)(ws + 591872);          // 2,048 B
  bf16*  h1  = (bf16*) (ws + 593920);          // 64 MB
  bf16*  enc = (bf16*) (ws + 593920 + 67108864);
  float* part= (float*)(ws + 593920 + 2ull * 67108864);  // 24 MB

  prep_weights<<<576, 256, 0, stream>>>(c1w, g1, b1, m1, v1,
                                        c2w, g2, b2, m2, v2,
                                        wf1, wf2, sh1, sh2);
  conv_block<false><<<dim3(8, 16, 16), 256, 0, stream>>>(x, (const bf16*)nullptr,
                                                         wf1, sh1, h1);
  conv_block<true ><<<dim3(8, 16, 16), 256, 0, stream>>>(x, h1, wf2, sh2, enc);
  proj_gemm<<<dim3(48, 8), 128, 0, stream>>>(enc, pw, part);
  reduce_bias<<<768, 256, 0, stream>>>(part, pb, (float*)d_out);
}
